// MultiHeadAttention_2_47261820125773
// MI455X (gfx1250) — matrix operand order repaired and checked by reading, not compiled, not run on hardware
//
#include <hip/hip_runtime.h>

typedef __attribute__((ext_vector_type(16))) _Float16 v16h;
typedef __attribute__((ext_vector_type(8)))  _Float16 v8h;
typedef __attribute__((ext_vector_type(8)))  float    v8f;

#define D_MODEL 1024
#define SEQ     2048
#define BATCH   2
#define NHEAD   16
#define DHEAD   64
#define MROWS   (BATCH*SEQ)   // 4096

union AF { v16h v; v8h h[2]; };

__device__ __forceinline__ v8f wmma_f16(v16h a, v16h b, v8f c) {
    return __builtin_amdgcn_wmma_f32_16x16x32_f16(
        false, a, false, b, (short)0, c, false, false);
}

// byte offset of an LDS object (low 32 bits of the shared-aperture address)
__device__ __forceinline__ unsigned lds_off(const void* p) {
    return (unsigned)(unsigned long long)p;
}

// async copy 16B: LDS[ldsoff] <= MEM[base + goff]   (GVS mode, ASYNCcnt)
__device__ __forceinline__ void async_cp16(unsigned ldsoff, const void* base, unsigned goff) {
    asm volatile("global_load_async_to_lds_b128 %0, %1, %2"
                 :: "v"(ldsoff), "v"(goff), "s"(base) : "memory");
}
#define WAIT_ASYNC(n) asm volatile("s_wait_asynccnt " #n ::: "memory")

// ---------------- conversion kernels ----------------

__global__ void k_cvt(const float* __restrict__ in, _Float16* __restrict__ out, int n) {
    int i = blockIdx.x * blockDim.x + threadIdx.x;
    if (i < n) out[i] = (_Float16)in[i];
}

// out[n*K + k] = in[k*N + n]  (f32 [K x N] -> f16 [N x K])
__global__ void k_cvt_t(const float* __restrict__ in, _Float16* __restrict__ out, int K, int N) {
    int i = blockIdx.x * blockDim.x + threadIdx.x;
    if (i < K * N) {
        int n = i / K, k = i % K;
        out[i] = (_Float16)in[k * N + n];
    }
}

// stage a 128-row x 64-half panel (rows at stride ldg halfs) into LDS, async.
// 1024 16B segments, 256 threads -> 4 async ops per thread.
__device__ __forceinline__ void stage_panel64(const _Float16* __restrict__ G, int ldg,
                                              int row0, int k0, _Float16* lds, int t) {
    unsigned lbase = lds_off(lds);
#pragma unroll
    for (int i = 0; i < 4; ++i) {
        int c = i * 256 + t;               // 0..1023
        int row = c >> 3, seg = (c & 7) * 8;
        unsigned goff = (unsigned)(((row0 + row) * ldg + k0 + seg) * 2);
        async_cp16(lbase + (unsigned)((row * 64 + seg) * 2), G, goff);
    }
}

// ---------------- QKV projection GEMM (LDS-staged, double-buffered) ----------
// A: x_h [4096 x 1024] f16. Bt: W^T [N x K] f16.  z selects Q / K / V.
__global__ void __launch_bounds__(256)
k_gemm_qkv(const _Float16* __restrict__ A,
           const _Float16* __restrict__ Wqt, const _Float16* __restrict__ Wkt,
           const _Float16* __restrict__ Wvt,
           _Float16* __restrict__ Qo, _Float16* __restrict__ Ko, _Float16* __restrict__ Vo)
{
    __shared__ __align__(32) _Float16 ldsA[2][128 * 64];
    __shared__ __align__(32) _Float16 ldsB[2][128 * 64];
    const int t = threadIdx.x;
    const int lane = t & 31, wid = t >> 5;
    const int lr = lane & 15, lh = lane >> 4;
    const int wm = wid >> 2, wn = wid & 3;   // 2 x 4 wave grid
    const int m0 = blockIdx.x * 128;
    const int n0 = blockIdx.y * 128;

    const _Float16* Bt; _Float16* Out; int vmode;
    if (blockIdx.z == 0)      { Bt = Wqt; Out = Qo; vmode = 0; }
    else if (blockIdx.z == 1) { Bt = Wkt; Out = Ko; vmode = 0; }
    else                      { Bt = Wvt; Out = Vo; vmode = 1; }

    v8f acc[4][2] = {};

    stage_panel64(A,  D_MODEL, m0, 0, ldsA[0], t);
    stage_panel64(Bt, D_MODEL, n0, 0, ldsB[0], t);

    const int KT = D_MODEL / 64;
    for (int kt = 0; kt < KT; ++kt) {
        const int cur = kt & 1;
        if (kt + 1 < KT) {
            stage_panel64(A,  D_MODEL, m0, (kt + 1) * 64, ldsA[cur ^ 1], t);
            stage_panel64(Bt, D_MODEL, n0, (kt + 1) * 64, ldsB[cur ^ 1], t);
            WAIT_ASYNC(0x8);      // previous group (8 in-order ops) complete
        } else {
            WAIT_ASYNC(0x0);
        }
        __syncthreads();

#pragma unroll
        for (int ks = 0; ks < 2; ++ks) {
            AF a[4];
#pragma unroll
            for (int mi = 0; mi < 4; ++mi) {
                const _Float16* p = &ldsA[cur][(wm * 64 + mi * 16 + lr) * 64 + ks * 32 + lh * 16];
                a[mi].h[0] = *(const v8h*)p;
                a[mi].h[1] = *(const v8h*)(p + 8);
            }
#pragma unroll
            for (int ni = 0; ni < 2; ++ni) {
                v16h b = *(const v16h*)&ldsB[cur][(wn * 32 + ni * 16 + lr) * 64 + ks * 32 + lh * 16];
#pragma unroll
                for (int mi = 0; mi < 4; ++mi)
                    acc[mi][ni] = wmma_f16(a[mi].v, b, acc[mi][ni]);
            }
        }
        __syncthreads();
    }

#pragma unroll
    for (int mi = 0; mi < 4; ++mi)
#pragma unroll
    for (int ni = 0; ni < 2; ++ni)
#pragma unroll
    for (int r = 0; r < 8; ++r) {
        int m = m0 + wm * 64 + mi * 16 + r + 8 * lh;   // b*S + s
        int n = n0 + wn * 32 + ni * 16 + lr;           // h*Dh + d
        int b = m >> 11, s = m & (SEQ - 1);
        int h = n >> 6,  d = n & (DHEAD - 1);
        int idx = (vmode == 0) ? (((b * NHEAD + h) * SEQ + s) * DHEAD + d)
                               : (((b * NHEAD + h) * DHEAD + d) * SEQ + s);
        Out[idx] = (_Float16)acc[mi][ni][r];
    }
}

// ---------------- streaming attention (LDS-staged K/V, online softmax) ------
// Block: 128 query rows of one (b,h); wave w owns 16 rows. Keys in chunks of 64.
__global__ void __launch_bounds__(256)
k_attn(const _Float16* __restrict__ Q, const _Float16* __restrict__ Kk,
       const _Float16* __restrict__ Vt, _Float16* __restrict__ Ctx)
{
    __shared__ __align__(32) _Float16 ldsK[2][64 * 64];  // 64 keys x 64 d
    __shared__ __align__(32) _Float16 ldsV[2][64 * 64];  // 64 d x 64 keys (Vt rows)
    __shared__ __align__(32) _Float16 plds[8][16 * 64];  // per-wave P tile (16 x 64 keys)
    const int t = threadIdx.x;
    const int lane = t & 31, wid = t >> 5;
    const int lr = lane & 15, lh = lane >> 4;
    const int bh = blockIdx.x >> 4;                   // 16 query blocks per head
    const int q0 = (blockIdx.x & 15) * 128 + wid * 16;

    const _Float16* qp = Q  + (long)bh * SEQ * DHEAD;
    const _Float16* kp = Kk + (long)bh * SEQ * DHEAD;
    const _Float16* vp = Vt + (long)bh * DHEAD * SEQ;

    AF aQ[2];
#pragma unroll
    for (int ks = 0; ks < 2; ++ks) {
        const _Float16* p = qp + (q0 + lr) * DHEAD + ks * 32 + lh * 16;
        aQ[ks].h[0] = *(const v8h*)p;
        aQ[ks].h[1] = *(const v8h*)(p + 8);
    }

    v8f outv[4] = {};
    float mrow[8], lrow[8];
#pragma unroll
    for (int r = 0; r < 8; ++r) { mrow[r] = -1e30f; lrow[r] = 0.f; }
    _Float16* myld = plds[wid];

    // K chunk: 64 rows x 128B -> 512 x 16B -> 2 ops/thread. V chunk same.
    auto stageK = [&](int key0, _Float16* dst) {
        unsigned lbase = lds_off(dst);
#pragma unroll
        for (int i = 0; i < 2; ++i) {
            int c = i * 256 + t;
            int row = c >> 3, seg = (c & 7) * 8;
            unsigned goff = (unsigned)(((key0 + row) * DHEAD + seg) * 2);
            async_cp16(lbase + (unsigned)((row * 64 + seg) * 2), kp, goff);
        }
    };
    auto stageV = [&](int key0, _Float16* dst) {
        unsigned lbase = lds_off(dst);
#pragma unroll
        for (int i = 0; i < 2; ++i) {
            int c = i * 256 + t;
            int row = c >> 3, seg = (c & 7) * 8;    // row = d, seg along keys
            unsigned goff = (unsigned)((row * SEQ + key0 + seg) * 2);
            async_cp16(lbase + (unsigned)((row * 64 + seg) * 2), vp, goff);
        }
    };

    stageK(0, ldsK[0]);
    stageV(0, ldsV[0]);

    const int NC = SEQ / 64;
    for (int c = 0; c < NC; ++c) {
        const int cur = c & 1;
        if (c + 1 < NC) {
            stageK((c + 1) * 64, ldsK[cur ^ 1]);
            stageV((c + 1) * 64, ldsV[cur ^ 1]);
            WAIT_ASYNC(0x4);
        } else {
            WAIT_ASYNC(0x0);
        }
        __syncthreads();

        // scores: 16 queries x 64 keys from LDS K tile
        v8f sc[4] = {};
#pragma unroll
        for (int nf = 0; nf < 4; ++nf)
#pragma unroll
        for (int ks = 0; ks < 2; ++ks) {
            v16h b = *(const v16h*)&ldsK[cur][(nf * 16 + lr) * 64 + ks * 32 + lh * 16];
            sc[nf] = wmma_f16(aQ[ks].v, b, sc[nf]);
        }

        // online softmax per row (row = r + 8*lh, col = lane%16)
        float alpha[8];
#pragma unroll
        for (int r = 0; r < 8; ++r) {
            float s0 = sc[0][r] * 0.125f;     // 1/sqrt(64)
            float s1 = sc[1][r] * 0.125f;
            float s2 = sc[2][r] * 0.125f;
            float s3 = sc[3][r] * 0.125f;
            float mx = fmaxf(fmaxf(s0, s1), fmaxf(s2, s3));
#pragma unroll
            for (int off = 1; off < 16; off <<= 1)
                mx = fmaxf(mx, __shfl_xor(mx, off, 16));
            float mnew = fmaxf(mrow[r], mx);
            float a = __expf(mrow[r] - mnew);
            mrow[r] = mnew; alpha[r] = a;
            float p0 = __expf(s0 - mnew), p1 = __expf(s1 - mnew);
            float p2 = __expf(s2 - mnew), p3 = __expf(s3 - mnew);
            float rs = (p0 + p1) + (p2 + p3);
#pragma unroll
            for (int off = 1; off < 16; off <<= 1)
                rs += __shfl_xor(rs, off, 16);
            lrow[r] = lrow[r] * a + rs;
            int row = r + 8 * lh;
            myld[row * 64 +      lr] = (_Float16)p0;
            myld[row * 64 + 16 + lr] = (_Float16)p1;
            myld[row * 64 + 32 + lr] = (_Float16)p2;
            myld[row * 64 + 48 + lr] = (_Float16)p3;
        }
#pragma unroll
        for (int nb = 0; nb < 4; ++nb)
#pragma unroll
        for (int r = 0; r < 8; ++r) outv[nb][r] *= alpha[r];

        __syncthreads();   // P tile C-layout writes -> A-layout reads
        AF pA[2];
#pragma unroll
        for (int ks = 0; ks < 2; ++ks) {
            pA[ks].h[0] = *(const v8h*)&myld[lr * 64 + ks * 32 + lh * 16];
            pA[ks].h[1] = *(const v8h*)&myld[lr * 64 + ks * 32 + lh * 16 + 8];
        }
#pragma unroll
        for (int nb = 0; nb < 4; ++nb)
#pragma unroll
        for (int ks = 0; ks < 2; ++ks) {
            v16h b = *(const v16h*)&ldsV[cur][(nb * 16 + lr) * 64 + ks * 32 + lh * 16];
            outv[nb] = wmma_f16(pA[ks].v, b, outv[nb]);
        }
        __syncthreads();   // before next chunk overwrites LDS
    }

    const int b = bh >> 4, h = bh & 15;
#pragma unroll
    for (int r = 0; r < 8; ++r) {
        float rinv = 1.0f / lrow[r];
        int m = b * SEQ + q0 + r + 8 * lh;
#pragma unroll
        for (int nb = 0; nb < 4; ++nb) {
            int n = h * DHEAD + nb * 16 + lr;
            Ctx[(long)m * D_MODEL + n] = (_Float16)(outv[nb][r] * rinv);
        }
    }
}

// ---------------- output projection GEMM (+bias, f32 out) -------------------
__global__ void __launch_bounds__(256)
k_gemm_out(const _Float16* __restrict__ A, const _Float16* __restrict__ Bt,
           const float* __restrict__ bias, float* __restrict__ Out)
{
    __shared__ __align__(32) _Float16 ldsA[2][128 * 64];
    __shared__ __align__(32) _Float16 ldsB[2][128 * 64];
    const int t = threadIdx.x;
    const int lane = t & 31, wid = t >> 5;
    const int lr = lane & 15, lh = lane >> 4;
    const int wm = wid >> 2, wn = wid & 3;
    const int m0 = blockIdx.x * 128;
    const int n0 = blockIdx.y * 128;

    v8f acc[4][2] = {};
    stage_panel64(A,  D_MODEL, m0, 0, ldsA[0], t);
    stage_panel64(Bt, D_MODEL, n0, 0, ldsB[0], t);

    const int KT = D_MODEL / 64;
    for (int kt = 0; kt < KT; ++kt) {
        const int cur = kt & 1;
        if (kt + 1 < KT) {
            stage_panel64(A,  D_MODEL, m0, (kt + 1) * 64, ldsA[cur ^ 1], t);
            stage_panel64(Bt, D_MODEL, n0, (kt + 1) * 64, ldsB[cur ^ 1], t);
            WAIT_ASYNC(0x8);
        } else {
            WAIT_ASYNC(0x0);
        }
        __syncthreads();

#pragma unroll
        for (int ks = 0; ks < 2; ++ks) {
            AF a[4];
#pragma unroll
            for (int mi = 0; mi < 4; ++mi) {
                const _Float16* p = &ldsA[cur][(wm * 64 + mi * 16 + lr) * 64 + ks * 32 + lh * 16];
                a[mi].h[0] = *(const v8h*)p;
                a[mi].h[1] = *(const v8h*)(p + 8);
            }
#pragma unroll
            for (int ni = 0; ni < 2; ++ni) {
                v16h b = *(const v16h*)&ldsB[cur][(wn * 32 + ni * 16 + lr) * 64 + ks * 32 + lh * 16];
#pragma unroll
                for (int mi = 0; mi < 4; ++mi)
                    acc[mi][ni] = wmma_f16(a[mi].v, b, acc[mi][ni]);
            }
        }
        __syncthreads();
    }

#pragma unroll
    for (int mi = 0; mi < 4; ++mi)
#pragma unroll
    for (int ni = 0; ni < 2; ++ni)
#pragma unroll
    for (int r = 0; r < 8; ++r) {
        int m = m0 + wm * 64 + mi * 16 + r + 8 * lh;
        int n = n0 + wn * 32 + ni * 16 + lr;
        Out[(long)m * D_MODEL + n] = acc[mi][ni][r] + bias[n];
    }
}

// ---------------- host launcher ----------------
extern "C" void kernel_launch(void* const* d_in, const int* in_sizes, int n_in,
                              void* d_out, int out_size, void* d_ws, size_t ws_size,
                              hipStream_t stream)
{
    // setup_inputs() order: x, Wk, Wq, Wv, Wo, bo
    const float* x  = (const float*)d_in[0];
    const float* Wk = (const float*)d_in[1];
    const float* Wq = (const float*)d_in[2];
    const float* Wv = (const float*)d_in[3];
    const float* Wo = (const float*)d_in[4];
    const float* bo = (const float*)d_in[5];

    char* ws = (char*)d_ws;
    _Float16* x_h = (_Float16*)(ws);                      // 8 MB [4096 x 1024]
    _Float16* Wqt = (_Float16*)(ws + (size_t)( 8 << 20)); // 2 MB each, W^T
    _Float16* Wkt = (_Float16*)(ws + (size_t)(10 << 20));
    _Float16* Wvt = (_Float16*)(ws + (size_t)(12 << 20));
    _Float16* Wot = (_Float16*)(ws + (size_t)(14 << 20));
    _Float16* Qh  = (_Float16*)(ws + (size_t)(16 << 20)); // [B,H,S,Dh]
    _Float16* Kh  = (_Float16*)(ws + (size_t)(24 << 20)); // [B,H,S,Dh]
    _Float16* Vth = (_Float16*)(ws + (size_t)(32 << 20)); // [B,H,Dh,S]
    _Float16* Ch  = (_Float16*)(ws + (size_t)(40 << 20)); // [4096 x 1024]

    const int nx = MROWS * D_MODEL;
    const int nw = D_MODEL * D_MODEL;

    k_cvt<<<dim3((nx + 255) / 256), dim3(256), 0, stream>>>(x, x_h, nx);
    k_cvt_t<<<dim3((nw + 255) / 256), dim3(256), 0, stream>>>(Wq, Wqt, D_MODEL, D_MODEL);
    k_cvt_t<<<dim3((nw + 255) / 256), dim3(256), 0, stream>>>(Wk, Wkt, D_MODEL, D_MODEL);
    k_cvt_t<<<dim3((nw + 255) / 256), dim3(256), 0, stream>>>(Wv, Wvt, D_MODEL, D_MODEL);
    k_cvt_t<<<dim3((nw + 255) / 256), dim3(256), 0, stream>>>(Wo, Wot, D_MODEL, D_MODEL);

    k_gemm_qkv<<<dim3(MROWS / 128, D_MODEL / 128, 3), dim3(256), 0, stream>>>(
        x_h, Wqt, Wkt, Wvt, Qh, Kh, Vth);

    // B*H heads x (S/128) query blocks = 512 blocks
    k_attn<<<dim3(BATCH * NHEAD * (SEQ / 128)), dim3(256), 0, stream>>>(
        Qh, Kh, Vth, Ch);

    k_gemm_out<<<dim3(MROWS / 128, D_MODEL / 128), dim3(256), 0, stream>>>(
        Ch, Wot, bo, (float*)d_out);
}
